// CircuitGNN_60498909332080
// MI455X (gfx1250) — matrix operand order repaired and checked by reading, not compiled, not run on hardware
//
#include <hip/hip_runtime.h>
#include <math.h>

// ---------------- types for WMMA ----------------
typedef __attribute__((ext_vector_type(16))) __bf16 v16bf;
typedef __attribute__((ext_vector_type(8)))  float  v8f;
typedef __attribute__((ext_vector_type(4)))  unsigned int u32x4;

#define HEADS 8
#define CH 32
#define DIM 256

// sign-aware float atomic max via integer atomics (works with -inf init)
__device__ __forceinline__ void atomicMaxF(float* addr, float val) {
    if (val >= 0.0f) atomicMax((int*)addr, __float_as_int(val));
    else             atomicMin((unsigned int*)addr, __float_as_uint(val));
}

// ---------------- layer 0: fused q/k/v/skip scalar GEMM (K=13) ----------------
__global__ void k_layer0(const float* __restrict__ x,
                         const float* __restrict__ Wq, const float* __restrict__ bq,
                         const float* __restrict__ Wk, const float* __restrict__ bk,
                         const float* __restrict__ Wv, const float* __restrict__ bv,
                         const float* __restrict__ Ws, const float* __restrict__ bs,
                         float* __restrict__ q, float* __restrict__ k,
                         float* __restrict__ v, float* __restrict__ skip, int n) {
    int tid = blockIdx.x * blockDim.x + threadIdx.x;
    if (tid >= n * DIM) return;
    int i = tid >> 8, j = tid & 255;
    float xr[13];
#pragma unroll
    for (int t = 0; t < 13; ++t) xr[t] = x[i * 13 + t];
    float aq = bq[j], ak = bk[j], av = bv[j], as = bs[j];
#pragma unroll
    for (int t = 0; t < 13; ++t) {
        float xv = xr[t];
        aq += xv * Wq[t * DIM + j];
        ak += xv * Wk[t * DIM + j];
        av += xv * Wv[t * DIM + j];
        as += xv * Ws[t * DIM + j];
    }
    q[tid] = aq; k[tid] = ak; v[tid] = av; skip[tid] = as;
}

// ---------------- transpose+convert 4 weight mats (256x256) to col-major bf16 ----------------
__global__ void k_prep_w(const float* __restrict__ W0, const float* __restrict__ W1,
                         const float* __restrict__ W2, const float* __restrict__ W3,
                         __bf16* __restrict__ out) {
    int tid = blockIdx.x * blockDim.x + threadIdx.x;   // 4*65536
    if (tid >= 4 * 65536) return;
    int mat = tid >> 16, idx = tid & 65535;
    int ncol = idx >> 8, kk = idx & 255;
    const float* W = (mat == 0) ? W0 : (mat == 1) ? W1 : (mat == 2) ? W2 : W3;
    out[tid] = (__bf16)W[kk * DIM + ncol];             // out[mat][ncol*256 + kk]
}

// ---------------- WMMA bf16 GEMM, all 4 projections fused (A reused 4x) ----------------
// D_m = h(bf16)[n,256] @ W_m + bias_m for m in {q,k,v,skip}, fp32 out.
// grid: (n/16, 2); block 256 = 8 waves; each wave -> one 16x16 tile x 4 matrices.
__global__ void __launch_bounds__(256)
k_gemm_wmma4(const __bf16* __restrict__ hbf, const __bf16* __restrict__ Wt,
             const float* __restrict__ b0, const float* __restrict__ b1,
             const float* __restrict__ b2, const float* __restrict__ b3,
             float* __restrict__ o0, float* __restrict__ o1,
             float* __restrict__ o2, float* __restrict__ o3, int n) {
    int wave = threadIdx.x >> 5;
    int lane = threadIdx.x & 31;
    int laneHi = lane >> 4;            // 0 / 1
    int ln = lane & 15;
    int rowBase = blockIdx.x * 16;
    int colTile = blockIdx.y * 8 + wave;       // 0..15
    int col = colTile * 16 + ln;
    if (rowBase >= n) return;

    // A: row = rowBase+ln; elements 0..15 = K[kk + laneHi*16 .. +15]
    const u32x4* Arow = reinterpret_cast<const u32x4*>(hbf + (size_t)(rowBase + ln) * DIM);
    // B: col-major weights; elements 0..15 = K[kk + laneHi*16 .. +15]
    const u32x4* Bp = reinterpret_cast<const u32x4*>(Wt);

    union Pack { v16bf v; u32x4 u[2]; };
    Pack A, B0, B1, B2, B3;
    v8f acc0 = {0.f,0.f,0.f,0.f,0.f,0.f,0.f,0.f};
    v8f acc1 = acc0, acc2 = acc0, acc3 = acc0;

#pragma unroll
    for (int kk = 0; kk < DIM; kk += 32) {
        int ka = (kk + (laneHi << 4)) >> 3;                 // u32x4 (8 bf16) index
        A.u[0] = Arow[ka];
        A.u[1] = Arow[ka + 1];                              // +8 bf16
        int kb = (col * DIM + kk + (laneHi << 4)) >> 3;
        B0.u[0] = Bp[kb];          B0.u[1] = Bp[kb + 1];
        B1.u[0] = Bp[kb + 8192];   B1.u[1] = Bp[kb + 8193];    // +65536 bf16
        B2.u[0] = Bp[kb + 16384];  B2.u[1] = Bp[kb + 16385];
        B3.u[0] = Bp[kb + 24576];  B3.u[1] = Bp[kb + 24577];
        acc0 = __builtin_amdgcn_wmma_f32_16x16x32_bf16(false, A.v, false, B0.v, (short)0, acc0, false, false);
        acc1 = __builtin_amdgcn_wmma_f32_16x16x32_bf16(false, A.v, false, B1.v, (short)0, acc1, false, false);
        acc2 = __builtin_amdgcn_wmma_f32_16x16x32_bf16(false, A.v, false, B2.v, (short)0, acc2, false, false);
        acc3 = __builtin_amdgcn_wmma_f32_16x16x32_bf16(false, A.v, false, B3.v, (short)0, acc3, false, false);
    }

    float bv0 = b0[col], bv1 = b1[col], bv2 = b2[col], bv3 = b3[col];
#pragma unroll
    for (int r = 0; r < 8; ++r) {
        int row = rowBase + laneHi * 8 + r;
        if (row < n) {
            size_t o = (size_t)row * DIM + col;
            o0[o] = acc0[r] + bv0;
            o1[o] = acc1[r] + bv1;
            o2[o] = acc2[r] + bv2;
            o3[o] = acc3[r] + bv3;
        }
    }
}

// ---------------- softmax/agg pipeline ----------------
__global__ void k_init_soft(float* __restrict__ amax, float* __restrict__ denom,
                            float* __restrict__ aggU, int nh, int nd) {
    int tid = blockIdx.x * blockDim.x + threadIdx.x;
    if (tid < nh) { amax[tid] = -1e30f; denom[tid] = 0.0f; }
    if (tid < nd) aggU[tid] = 0.0f;
}

__global__ void k_alpha(const int* __restrict__ ei, const float* __restrict__ q,
                        const float* __restrict__ k, float* __restrict__ alpha,
                        float* __restrict__ amax, int nE) {
    int tid = blockIdx.x * blockDim.x + threadIdx.x;
    if (tid >= nE * HEADS) return;
    int e = tid >> 3, hh = tid & 7;
    int s = ei[e], d = ei[nE + e];
    const float4* qp = reinterpret_cast<const float4*>(q + (size_t)d * DIM + hh * CH);
    const float4* kp = reinterpret_cast<const float4*>(k + (size_t)s * DIM + hh * CH);
    float acc = 0.0f;
#pragma unroll
    for (int t = 0; t < 8; ++t) {
        float4 a = qp[t], b = kp[t];
        acc += a.x * b.x + a.y * b.y + a.z * b.z + a.w * b.w;
    }
    float al = acc * 0.1767766952966369f;   // 1/sqrt(32)
    alpha[tid] = al;
    atomicMaxF(&amax[d * HEADS + hh], al);
}

__global__ void k_agg(const int* __restrict__ ei, const float* __restrict__ alpha,
                      const float* __restrict__ amax, const float* __restrict__ v,
                      float* __restrict__ denom, float* __restrict__ aggU, int nE) {
    int tid = blockIdx.x * blockDim.x + threadIdx.x;
    if (tid >= nE * HEADS) return;
    int e = tid >> 3, hh = tid & 7;
    int s = ei[e], d = ei[nE + e];
    float ev = __expf(alpha[tid] - amax[d * HEADS + hh]);
    atomicAdd(&denom[d * HEADS + hh], ev);
    const float* vp = v + (size_t)s * DIM + hh * CH;
    float* op = aggU + (size_t)d * DIM + hh * CH;
#pragma unroll
    for (int c = 0; c < CH; ++c) atomicAdd(&op[c], ev * vp[c]);
}

// finalize: h = relu(skip + agg/denom); also emit bf16 copy for next layer's WMMA A operand
__global__ void k_finalize(const float* __restrict__ aggU, const float* __restrict__ denom,
                           float* __restrict__ hio, __bf16* __restrict__ hbf, int n) {
    int tid = blockIdx.x * blockDim.x + threadIdx.x;
    if (tid >= n * DIM) return;
    int i = tid >> 8, j = tid & 255;
    float r = hio[tid] + aggU[tid] / (denom[i * HEADS + (j >> 5)] + 1e-16f);
    r = (r > 0.0f) ? r : 0.0f;
    hio[tid] = r;
    hbf[tid] = (__bf16)r;
}

// ---------------- pooling + MLP + regressor ----------------
__global__ void k_zero_pool(float* __restrict__ pooled, float* __restrict__ counts, int G) {
    int tid = blockIdx.x * blockDim.x + threadIdx.x;
    if (tid < G * DIM) pooled[tid] = 0.0f;
    if (tid < G) counts[tid] = 0.0f;
}

__global__ void k_pool(const float* __restrict__ h, const int* __restrict__ batch,
                       float* __restrict__ pooled, float* __restrict__ counts, int n) {
    int tid = blockIdx.x * blockDim.x + threadIdx.x;
    if (tid >= n * DIM) return;
    int i = tid >> 8, j = tid & 255;
    int g = batch[i];
    atomicAdd(&pooled[g * DIM + j], h[tid]);
    if (j == 0) atomicAdd(&counts[g], 1.0f);
}

__global__ void k_mlp(const float* __restrict__ gf,
                      const float* __restrict__ W1, const float* __restrict__ b1,
                      const float* __restrict__ W2, const float* __restrict__ b2,
                      const float* __restrict__ W3, const float* __restrict__ b3,
                      float* __restrict__ gout, int G) {
    int g = blockIdx.x * blockDim.x + threadIdx.x;
    if (g >= G) return;
    float a0[8], a1[16], a2[16];
#pragma unroll
    for (int t = 0; t < 8; ++t) a0[t] = gf[g * 8 + t];
    for (int j = 0; j < 16; ++j) {
        float s = b1[j];
        for (int t = 0; t < 8; ++t) s += a0[t] * W1[t * 16 + j];
        a1[j] = fmaxf(s, 0.0f);
    }
    for (int j = 0; j < 16; ++j) {
        float s = b2[j];
        for (int t = 0; t < 16; ++t) s += a1[t] * W2[t * 16 + j];
        a2[j] = fmaxf(s, 0.0f);
    }
    for (int j = 0; j < 16; ++j) {
        float s = b3[j];
        for (int t = 0; t < 16; ++t) s += a2[t] * W3[t * 16 + j];
        gout[g * 16 + j] = fmaxf(s, 0.0f);
    }
}

__global__ void k_reg(const float* __restrict__ pooled, const float* __restrict__ counts,
                      const float* __restrict__ gvec,
                      const float* __restrict__ W1, const float* __restrict__ b1,
                      const float* __restrict__ W2, const float* __restrict__ b2,
                      const float* __restrict__ W3, const float* __restrict__ b3,
                      float* __restrict__ out, int G) {
    int g = blockIdx.x * blockDim.x + threadIdx.x;
    if (g >= G) return;
    float inv = 1.0f / fmaxf(counts[g], 1.0f);
    float a1[16], a2[16];
    for (int j = 0; j < 16; ++j) {
        float s = b1[j];
        for (int t = 0; t < DIM; ++t) s += (pooled[g * DIM + t] * inv) * W1[t * 16 + j];
        for (int t = 0; t < 16; ++t) s += gvec[g * 16 + t] * W1[(DIM + t) * 16 + j];
        a1[j] = fmaxf(s, 0.0f);
    }
    for (int j = 0; j < 16; ++j) {
        float s = b2[j];
        for (int t = 0; t < 16; ++t) s += a1[t] * W2[t * 16 + j];
        a2[j] = fmaxf(s, 0.0f);
    }
    float s = b3[0];
    for (int t = 0; t < 16; ++t) s += a2[t] * W3[t];
    out[g] = s;
}

// ---------------- host side ----------------
static inline int nblk(long n) { return (int)((n + 255) / 256); }

extern "C" void kernel_launch(void* const* d_in, const int* in_sizes, int n_in,
                              void* d_out, int out_size, void* d_ws, size_t ws_size,
                              hipStream_t stream) {
    const float* x     = (const float*)d_in[0];
    const int*   ei    = (const int*)d_in[1];
    const int*   batch = (const int*)d_in[2];
    const float* gf    = (const float*)d_in[3];
    const int N = in_sizes[2];
    const int E = in_sizes[1] / 2;
    const int G = in_sizes[3] / 8;

    const float* cw[5][8];
    for (int l = 0; l < 5; ++l)
        for (int t = 0; t < 8; ++t) cw[l][t] = (const float*)d_in[4 + l * 8 + t];
    const float* mlp[6];
    for (int t = 0; t < 6; ++t) mlp[t] = (const float*)d_in[44 + t];
    const float* reg[6];
    for (int t = 0; t < 6; ++t) reg[t] = (const float*)d_in[50 + t];

    // workspace carve (256B aligned)
    char* p = (char*)d_ws;
    auto carve = [&](size_t bytes) -> void* {
        void* r = (void*)p;
        p += (bytes + 255) & ~size_t(255);
        return r;
    };
    size_t ND = (size_t)N * DIM;
    float*  hA    = (float*)carve(ND * 4);
    float*  hB    = (float*)carve(ND * 4);
    float*  q     = (float*)carve(ND * 4);
    float*  k     = (float*)carve(ND * 4);
    float*  v     = (float*)carve(ND * 4);
    float*  aggU  = (float*)carve(ND * 4);
    __bf16* hbf   = (__bf16*)carve(ND * 2);
    float*  alpha = (float*)carve((size_t)E * HEADS * 4);
    float*  amax  = (float*)carve((size_t)N * HEADS * 4);
    float*  denom = (float*)carve((size_t)N * HEADS * 4);
    __bf16* Wt    = (__bf16*)carve((size_t)4 * 65536 * 2);
    float*  pooled = (float*)carve((size_t)G * DIM * 4);
    float*  counts = (float*)carve((size_t)G * 4);
    float*  gvec   = (float*)carve((size_t)G * 16 * 4);
    (void)ws_size; (void)n_in; (void)out_size;

    long EH = (long)E * HEADS;

    // ---- layer 0 (scalar, K=13); skip written into hA; finalize also emits bf16 ----
    k_layer0<<<nblk(ND), 256, 0, stream>>>(x, cw[0][0], cw[0][1], cw[0][2], cw[0][3],
                                           cw[0][4], cw[0][5], cw[0][6], cw[0][7],
                                           q, k, v, hA, N);
    k_init_soft<<<nblk(ND), 256, 0, stream>>>(amax, denom, aggU, N * HEADS, (int)ND);
    k_alpha<<<nblk(EH), 256, 0, stream>>>(ei, q, k, alpha, amax, E);
    k_agg<<<nblk(EH), 256, 0, stream>>>(ei, alpha, amax, v, denom, aggU, E);
    k_finalize<<<nblk(ND), 256, 0, stream>>>(aggU, denom, hA, hbf, N);

    // ---- layers 1..4 (WMMA, 4 projections fused) ----
    float* hin = hA;
    float* hout = hB;
    for (int l = 1; l < 5; ++l) {
        k_prep_w<<<nblk(4 * 65536), 256, 0, stream>>>(cw[l][0], cw[l][2], cw[l][4], cw[l][6], Wt);
        dim3 gg((N + 15) / 16, 2);
        k_gemm_wmma4<<<gg, 256, 0, stream>>>(hbf, Wt,
                                             cw[l][1], cw[l][3], cw[l][5], cw[l][7],
                                             q, k, v, hout, N);
        k_init_soft<<<nblk(ND), 256, 0, stream>>>(amax, denom, aggU, N * HEADS, (int)ND);
        k_alpha<<<nblk(EH), 256, 0, stream>>>(ei, q, k, alpha, amax, E);
        k_agg<<<nblk(EH), 256, 0, stream>>>(ei, alpha, amax, v, denom, aggU, E);
        k_finalize<<<nblk(ND), 256, 0, stream>>>(aggU, denom, hout, hbf, N);
        float* t = hin; hin = hout; hout = t;
    }
    // final node features in hin (fp32)

    // ---- global mean pool + MLP + regressor ----
    k_zero_pool<<<nblk((long)G * DIM), 256, 0, stream>>>(pooled, counts, G);
    k_pool<<<nblk(ND), 256, 0, stream>>>(hin, batch, pooled, counts, N);
    k_mlp<<<nblk(G), 256, 0, stream>>>(gf, mlp[0], mlp[1], mlp[2], mlp[3], mlp[4], mlp[5], gvec, G);
    k_reg<<<nblk(G), 256, 0, stream>>>(pooled, counts, gvec,
                                       reg[0], reg[1], reg[2], reg[3], reg[4], reg[5],
                                       (float*)d_out, G);
}
